// self_LSTM_sparse_attn_predict_25357486916380
// MI455X (gfx1250) — matrix operand order repaired and checked by reading, not compiled, not run on hardware
//
#include <hip/hip_runtime.h>
#include <hip/hip_bf16.h>

// ---------------------------------------------------------------------------
// Sizes from the reference
// ---------------------------------------------------------------------------
#define Bdim   32
#define Tdim   128
#define Ddim   128
#define Hdim   512
#define G4H    2048          // 4*H
#define TOPK   5
#define EPSC   1e-7f
#define NEGC   (-1.0e9f)

typedef __attribute__((ext_vector_type(16))) _Float16 v16h;
typedef __attribute__((ext_vector_type(8)))  _Float16 v8h;
typedef __attribute__((ext_vector_type(8)))  float    v8f;

union AFrag { v16h v; v8h h[2]; };

__device__ __forceinline__ float sigmoidf_(float x) { return 1.0f / (1.0f + expf(-x)); }

// ---------------------------------------------------------------------------
// Init: f32 -> f16 conversions of x / W_ih / W_hh, zero h_f16, c, S2, Hold.
// ---------------------------------------------------------------------------
__global__ void k_init(const float* __restrict__ x, const float* __restrict__ Wih,
                       const float* __restrict__ Whh,
                       _Float16* __restrict__ x16, _Float16* __restrict__ Wih16,
                       _Float16* __restrict__ Whh16, _Float16* __restrict__ hf16,
                       float* __restrict__ cbuf, float* __restrict__ S2,
                       float* __restrict__ Hold)
{
    const int idx    = blockIdx.x * blockDim.x + threadIdx.x;
    const int stride = gridDim.x * blockDim.x;
    for (int i = idx; i < Bdim * Tdim * Ddim; i += stride) x16[i]   = (_Float16)x[i];
    for (int i = idx; i < G4H * Ddim;         i += stride) Wih16[i] = (_Float16)Wih[i];
    for (int i = idx; i < G4H * Hdim;         i += stride) Whh16[i] = (_Float16)Whh[i];
    for (int i = idx; i < Bdim * Hdim;        i += stride) { hf16[i] = (_Float16)0.0f; cbuf[i] = 0.0f; }
    for (int i = idx; i < Bdim * Tdim;        i += stride) S2[i] = 0.0f;
    for (int i = idx; i < Bdim * Tdim * Hdim; i += stride) Hold[i] = 0.0f;
}

// ---------------------------------------------------------------------------
// Hoisted GEMM: Gx[t][b][n] = (x @ W_ih^T)[b*T+t][n] + b_ih[n] + b_hh[n]
// A = x16 (4096 x 128), B[k][n] = Wih16[n*128+k].  8 waves / block, 1 tile / wave.
// ---------------------------------------------------------------------------
__global__ void k_gemm_pre(const _Float16* __restrict__ A, const _Float16* __restrict__ Bt,
                           const float* __restrict__ b_ih, const float* __restrict__ b_hh,
                           float* __restrict__ Gx)
{
    const int lane  = threadIdx.x & 31;
    const int wid   = threadIdx.x >> 5;
    const int idx   = blockIdx.x * 8 + wid;     // tile id
    const int n0    = (idx & 127) * 16;
    const int m0    = (idx >> 7)  * 16;
    const int half  = lane >> 4;
    const int l15   = lane & 15;

    v8f c = {};
    const _Float16* arow = A  + (size_t)(m0 + l15) * Ddim;
    const _Float16* brow = Bt + (size_t)(n0 + l15) * Ddim;

#pragma unroll
    for (int k0 = 0; k0 < Ddim; k0 += 32) {
        AFrag a;
        a.h[0] = *(const v8h*)(arow + k0 + 8 * half);       // K = k0 + 8*half .. +7
        a.h[1] = *(const v8h*)(arow + k0 + 16 + 8 * half);  // K = k0+16+8*half .. +7
        AFrag b;
        b.h[0] = *(const v8h*)(brow + k0 + 8 * half);       // K = k0 + 8*half .. +7
        b.h[1] = *(const v8h*)(brow + k0 + 16 + 8 * half);  // K = k0+16+8*half .. +7
        c = __builtin_amdgcn_wmma_f32_16x16x32_f16(false, a.v, false, b.v, (short)0, c, false, false);
    }
#pragma unroll
    for (int j = 0; j < 8; ++j) {
        const int m  = m0 + j + 8 * half;       // row in (B*T)
        const int bb = m >> 7;                  // T = 128
        const int t  = m & 127;
        const int n  = n0 + l15;
        Gx[((size_t)t * Bdim + bb) * G4H + n] = c[j] + b_ih[n] + b_hh[n];
    }
}

// ---------------------------------------------------------------------------
// Per-step GEMM: gates = Gx_t + h @ W_hh^T.
// 8 waves / block share one 16x512 f16 A tile, staged into LDS via the CDNA5
// async global->LDS path (ASYNCcnt).  Two accumulator chains hide WMMA latency.
// ---------------------------------------------------------------------------
__global__ void k_gemm_step(const _Float16* __restrict__ A, const _Float16* __restrict__ Bt,
                            const float* __restrict__ Cin, float* __restrict__ gates)
{
    const int lane = threadIdx.x & 31;
    const int wid  = threadIdx.x >> 5;
    const int idx  = blockIdx.x * 8 + wid;      // tile id, 0..255
    const int n0   = (idx & 127) * 16;
    const int m0   = (idx >> 7)  * 16;          // same for all 8 waves in a block
    const int half = lane >> 4;
    const int l15  = lane & 15;

    __shared__ __align__(32) _Float16 As[16 * Hdim];   // 16 KB staged A tile

    // Async copy A tile (h rows m0..m0+15) global -> LDS: 1024 x 16B chunks.
    {
        const int tid = threadIdx.x;
        const _Float16* Abase = A + (size_t)m0 * Hdim;
#pragma unroll
        for (int p = 0; p < 4; ++p) {
            const int c    = tid + p * 256;     // 16B chunk index (64 per row)
            const int row  = c >> 6;
            const int col8 = (c & 63) * 8;      // halves
            const uint64_t g = (uint64_t)(uintptr_t)(Abase + (size_t)row * Hdim + col8);
            const uint32_t l = (uint32_t)(uintptr_t)(&As[row * Hdim + col8]);
            asm volatile("global_load_async_to_lds_b128 %0, %1, off"
                         :: "v"(l), "v"(g) : "memory");
        }
        asm volatile("s_wait_asynccnt 0x0" ::: "memory");
        __syncthreads();
    }

    v8f c0, c1 = {};
#pragma unroll
    for (int j = 0; j < 8; ++j)
        c0[j] = Cin[(size_t)(m0 + j + 8 * half) * G4H + n0 + l15];

    const _Float16* arow = &As[l15 * Hdim];                 // LDS, row = M
    const _Float16* brow = Bt + (size_t)(n0 + l15) * Hdim;  // global, col-of-B

#pragma unroll
    for (int k0 = 0; k0 < Hdim; k0 += 64) {
        AFrag a0, a1;
        a0.h[0] = *(const v8h*)(arow + k0 + 8 * half);
        a0.h[1] = *(const v8h*)(arow + k0 + 16 + 8 * half);
        a1.h[0] = *(const v8h*)(arow + k0 + 32 + 8 * half);
        a1.h[1] = *(const v8h*)(arow + k0 + 48 + 8 * half);
        AFrag b0, b1;
        b0.h[0] = *(const v8h*)(brow + k0 + 8 * half);
        b0.h[1] = *(const v8h*)(brow + k0 + 16 + 8 * half);
        b1.h[0] = *(const v8h*)(brow + k0 + 32 + 8 * half);
        b1.h[1] = *(const v8h*)(brow + k0 + 48 + 8 * half);
        c0 = __builtin_amdgcn_wmma_f32_16x16x32_f16(false, a0.v, false, b0.v, (short)0, c0, false, false);
        c1 = __builtin_amdgcn_wmma_f32_16x16x32_f16(false, a1.v, false, b1.v, (short)0, c1, false, false);
    }
#pragma unroll
    for (int j = 0; j < 8; ++j)
        gates[(size_t)(m0 + j + 8 * half) * G4H + n0 + l15] = c0[j] + c1[j];
}

// ---------------------------------------------------------------------------
// Fused per-step tail (one block per batch row b):
//   LSTM cell -> s1 reduction -> scores + top-5 -> weights -> attn_c ->
//   h = h_lstm + attn_c -> write Hold row r, h_f16, S2[b,r].
// ---------------------------------------------------------------------------
__global__ void k_step_tail(const float* __restrict__ gates, float* __restrict__ cbuf,
                            float* __restrict__ Hold, const float* __restrict__ w_t,
                            _Float16* __restrict__ hf16, float* __restrict__ S2,
                            float* __restrict__ out_c, float* __restrict__ out_w, int r)
{
    const int b = blockIdx.x;
    const int h = threadIdx.x;                  // 512 threads
    const float* g = gates + (size_t)b * G4H;

    // ---- LSTM cell (h_lstm kept in register) ----
    const float ig = sigmoidf_(g[h]);
    const float fg = sigmoidf_(g[Hdim + h]);
    const float gg = tanhf    (g[2 * Hdim + h]);
    const float og = sigmoidf_(g[3 * Hdim + h]);
    const float cn = fg * cbuf[(size_t)b * Hdim + h] + ig * gg;
    const float hl = og * tanhf(cn);
    cbuf[(size_t)b * Hdim + h] = cn;

    __shared__ float red[Hdim];
    __shared__ float skey[Tdim];
    __shared__ float sv[Tdim];
    __shared__ int   si[Tdim];
    __shared__ float wsh[Tdim];
    __shared__ float sscal[2];                  // [0]=s1, [1]=5th-max

    // ---- s1[b] = sum tanh(h_lstm)*w1 ----
    red[h] = tanhf(hl) * w_t[h];
    __syncthreads();
    for (int s = Hdim / 2; s > 0; s >>= 1) {
        if (h < s) red[h] += red[h + s];
        __syncthreads();
    }
    if (h == 0) sscal[0] = red[0];
    __syncthreads();

    // ---- scores ----
    float s = NEGC;
    if (h < Tdim) {
        s = (h < r) ? (sscal[0] + S2[(size_t)b * Tdim + h]) : NEGC;
        skey[h] = s;
    }
    __syncthreads();

    // ---- 5 serial argmax passes -> 5th largest ----
    for (int it = 0; it < TOPK; ++it) {
        if (h < Tdim) { sv[h] = skey[h]; si[h] = h; }
        __syncthreads();
        for (int st = Tdim / 2; st > 0; st >>= 1) {
            if (h < st && sv[h + st] > sv[h]) { sv[h] = sv[h + st]; si[h] = si[h + st]; }
            __syncthreads();
        }
        if (h == 0) { sscal[1] = sv[0]; skey[si[0]] = -3.0e38f; }
        __syncthreads();
    }
    const float delta = sscal[1] + EPSC;

    // ---- weights ----
    if (h < Tdim) sv[h] = fmaxf(s - delta, 0.0f);
    __syncthreads();
    for (int st = Tdim / 2; st > 0; st >>= 1) {
        if (h < st) sv[h] += sv[h + st];
        __syncthreads();
    }
    if (h < Tdim) {
        const float aw  = fmaxf(s - delta, 0.0f) / (sv[0] + EPSC);
        const float raw = (h < r) ? s : 0.0f;
        const float wf  = (r <= TOPK) ? raw : aw;
        wsh[h] = wf;
        out_w[(size_t)b * Tdim + h] = wf;       // reference output attn_w (B,T,1)
    }
    __syncthreads();

    // ---- attn_c and state update ----
    const float* Hb = Hold + (size_t)b * Tdim * Hdim;
    float acc = 0.0f;
    for (int t = 0; t < r; ++t)                 // weights are 0 for invalid t
        acc += wsh[t] * Hb[(size_t)t * Hdim + h];

    const float hn = hl + acc;
    out_c[(size_t)b * Hdim + h] = acc;          // reference output attn_c (B,H)
    hf16 [(size_t)b * Hdim + h] = (_Float16)hn;
    if (r < Tdim) Hold[(size_t)b * Tdim * Hdim + (size_t)r * Hdim + h] = hn;

    // ---- S2[b,r] = sum tanh(h)*w2 ----
    red[h] = tanhf(hn) * w_t[Hdim + h];
    __syncthreads();
    for (int ss = Hdim / 2; ss > 0; ss >>= 1) {
        if (h < ss) red[h] += red[h + ss];
        __syncthreads();
    }
    if (h == 0 && r < Tdim) S2[(size_t)b * Tdim + r] = red[0];
}

// ---------------------------------------------------------------------------
// Host launcher
// ---------------------------------------------------------------------------
extern "C" void kernel_launch(void* const* d_in, const int* in_sizes, int n_in,
                              void* d_out, int out_size, void* d_ws, size_t ws_size,
                              hipStream_t stream)
{
    const float* x    = (const float*)d_in[0];
    const float* Wih  = (const float*)d_in[1];
    const float* Whh  = (const float*)d_in[2];
    const float* b_ih = (const float*)d_in[3];
    const float* b_hh = (const float*)d_in[4];
    const float* w_t  = (const float*)d_in[5];

    float* out_c = (float*)d_out;                 // attn_c: 32*512
    float* out_w = (float*)d_out + Bdim * Hdim;   // attn_w: 32*128

    uint8_t* ws  = (uint8_t*)d_ws;
    size_t   off = 0;
    auto alloc = [&](size_t bytes) -> void* {
        void* p = ws + off;
        off = (off + bytes + 255) & ~(size_t)255;
        return p;
    };

    _Float16* x16   = (_Float16*)alloc((size_t)Bdim * Tdim * Ddim * 2);
    _Float16* Wih16 = (_Float16*)alloc((size_t)G4H * Ddim * 2);
    _Float16* Whh16 = (_Float16*)alloc((size_t)G4H * Hdim * 2);
    float*    Gx    = (float*)   alloc((size_t)Bdim * Tdim * G4H * 4);   // 32 MB, L2-resident
    float*    gates = (float*)   alloc((size_t)Bdim * G4H * 4);
    float*    cbuf  = (float*)   alloc((size_t)Bdim * Hdim * 4);
    _Float16* hf16  = (_Float16*)alloc((size_t)Bdim * Hdim * 2);
    float*    Hold  = (float*)   alloc((size_t)Bdim * Tdim * Hdim * 4);  // 8 MB
    float*    S2    = (float*)   alloc((size_t)Bdim * Tdim * 4);

    k_init<<<2048, 256, 0, stream>>>(x, Wih, Whh, x16, Wih16, Whh16, hf16, cbuf, S2, Hold);

    // Hoisted x-GEMM: 32768 tiles, 8 waves per block.
    k_gemm_pre<<<4096, 256, 0, stream>>>(x16, Wih16, b_ih, b_hh, Gx);

    for (int i = 0; i < Tdim; ++i) {
        const int r = i + 1;
        k_gemm_step<<<32, 256, 0, stream>>>(hf16, Whh16, Gx + (size_t)i * Bdim * G4H, gates);
        k_step_tail<<<Bdim, Hdim, 0, stream>>>(gates, cbuf, Hold, w_t, hf16, S2, out_c, out_w, r);
    }
}
